// AnchorOnlyMixtureRNN_40656160424378
// MI455X (gfx1250) — matrix operand order repaired and checked by reading, not compiled, not run on hardware
//
#include <hip/hip_runtime.h>
#include <hip/hip_bf16.h>
#include <math.h>

// ---------------------------------------------------------------------------
// AnchorOnlyMixtureRNN for MI455X (gfx1250, wave32, WMMA).
//
// Algebraic refactor: serial recurrences carry ONLY z (LayerNorm scans);
// every matmul is hoisted into large batched WMMA GEMMs:
//   W~ = Qa@Wk   (64x512)      gate = sigmoid(scale*(Zhist@W~^T + c))  [32768x64]
//   K~ = Kr@Wq   (2048x512)    att  = scale*(z . K~[b,a] + cb[b,a])
// Serial parts (encoder z-scan, decoder attention scan) are 32 independent
// single-wave blocks: LN via wave32 shuffles, embedding prefetch 8 ahead.
// ---------------------------------------------------------------------------

typedef _Float16 h16;
typedef __attribute__((ext_vector_type(16))) _Float16 v16h;
typedef __attribute__((ext_vector_type(8)))  float    v8f;

#define DD   512
#define AA   64
#define BB   32
#define SENC 1024
#define SDEC 256
#define VOUT 32000
#define LN_EPS 1e-6f
#define SCALE 0.04419417382415922f   /* 1/sqrt(512) */
#define SQRTD 22.627416997969522f    /* sqrt(512)   */

__device__ __forceinline__ float wave_sum(float v) {
#pragma unroll
  for (int o = 16; o > 0; o >>= 1) v += __shfl_xor(v, o);
  return v;
}
__device__ __forceinline__ float wave_max(float v) {
#pragma unroll
  for (int o = 16; o > 0; o >>= 1) v = fmaxf(v, __shfl_xor(v, o));
  return v;
}

__device__ __forceinline__ v8f wmma16(v16h a, v16h b, v8f c) {
  return __builtin_amdgcn_wmma_f32_16x16x32_f16(false, a, false, b,
                                                (short)0, c, false, false);
}

// A fragment 16x32 f16 (row-major source, row stride ld halves):
// lane l -> row m=l&15, halves [c..c+15], c=(l>=16)*16.
__device__ __forceinline__ v16h load_a(const h16* __restrict__ src, int lane,
                                       int kk, int ld) {
  const int m = lane & 15;
  const int c = (lane >> 4) << 4;
  const h16* p = src + (size_t)m * ld + kk + c;
  v16h a;
#pragma unroll
  for (int j = 0; j < 8; ++j) a[j] = p[j];
#pragma unroll
  for (int j = 0; j < 8; ++j) a[8 + j] = p[8 + j];
  return a;
}
// B fragment 32x16 with B[k][n] = W[n][k]; 32B contiguous run from W row.
__device__ __forceinline__ v16h load_bfrag(const h16* __restrict__ W, int lane,
                                           int nb, int kk, int ld) {
  const int n = nb + (lane & 15);
  const int c = (lane >> 4) << 4;
  const h16* p = W + (size_t)n * ld + kk + c;
  v16h b;
#pragma unroll
  for (int j = 0; j < 16; ++j) b[j] = p[j];
  return b;
}
__device__ __forceinline__ v16h load_bfrag(const float* __restrict__ W, int lane,
                                           int nb, int kk, int ld) {
  const int n = nb + (lane & 15);
  const int c = (lane >> 4) << 4;
  const float* p = W + (size_t)n * ld + kk + c;
  v16h b;
#pragma unroll
  for (int j = 0; j < 16; ++j) b[j] = (h16)p[j];
  return b;
}

// ---------------------------------------------------------------- prep ----
__global__ void __launch_bounds__(256)
k_transpose_to_f16(const float* __restrict__ src, h16* __restrict__ dst) {
  const int idx = blockIdx.x * 256 + threadIdx.x;     // 512*512
  const int i = idx >> 9, j = idx & (DD - 1);
  dst[(size_t)j * DD + i] = (h16)src[(size_t)i * DD + j];
}

// Qa[a,d] = sum_k keyW[a,k]*Wq[d,k] + bq[d]
__global__ void __launch_bounds__(256)
k_qa(const float* __restrict__ keyW, const float* __restrict__ Wq,
     const float* __restrict__ bq, h16* __restrict__ Qa16,
     float* __restrict__ Qaf) {
  const int idx = blockIdx.x * 256 + threadIdx.x;     // A*D
  const int a = idx >> 9, d = idx & (DD - 1);
  const float* ka = keyW + (size_t)a * DD;
  const float* wd = Wq + (size_t)d * DD;
  float s = bq[d];
#pragma unroll 8
  for (int k = 0; k < DD; ++k) s += ka[k] * wd[k];
  Qaf[(size_t)a * DD + d] = s;
  Qa16[(size_t)a * DD + d] = (h16)s;
}

// cg[a] = sum_n bk[n]*Qa[a,n]
__global__ void __launch_bounds__(64)
k_cvec(const float* __restrict__ bk, const float* __restrict__ Qaf,
       float* __restrict__ cg) {
  const int a = threadIdx.x;
  const float* qa = Qaf + (size_t)a * DD;
  float s = 0.f;
#pragma unroll 8
  for (int n = 0; n < DD; ++n) s += bk[n] * qa[n];
  cg[a] = s;
}

// cb[m] = sum_n bq[n]*Kr16[m,n]
__global__ void __launch_bounds__(256)
k_cb(const float* __restrict__ bq, const h16* __restrict__ Kr16,
     float* __restrict__ cb) {
  const int m = blockIdx.x * 256 + threadIdx.x;       // < 2048
  const h16* kr = Kr16 + (size_t)m * DD;
  float s = 0.f;
#pragma unroll 8
  for (int n = 0; n < DD; ++n) s += bq[n] * (float)kr[n];
  cb[m] = s;
}

// ------------------------------------------ encoder z-scan (pure VALU) ----
// 32 independent single-wave blocks; token-row prefetch 8 steps ahead.
__global__ void __launch_bounds__(32)
k_enc_zscan(const int* __restrict__ ids, const float* __restrict__ emb_in,
            const float* __restrict__ n1g, const float* __restrict__ n1b,
            h16* __restrict__ z16hist, float* __restrict__ zfin) {
  const int b = blockIdx.x;
  const int lane = threadIdx.x;
  const int d0 = lane << 4;
  float zr[16], g1[16], b1[16];
#pragma unroll
  for (int j = 0; j < 16; ++j) {
    zr[j] = 0.f;
    g1[j] = n1g[d0 + j];
    b1[j] = n1b[d0 + j];
  }
#pragma unroll
  for (int t = 0; t < 8; ++t) {
    const int tk = ids[(size_t)b * SENC + t];
    __builtin_prefetch(emb_in + (size_t)tk * DD + d0, 0, 0);
  }
  for (int t = 0; t < SENC; ++t) {
    if (t + 8 < SENC) {
      const int tk = ids[(size_t)b * SENC + t + 8];
      __builtin_prefetch(emb_in + (size_t)tk * DD + d0, 0, 0);
    }
    const int tok = ids[(size_t)b * SENC + t];
    const float* e = emb_in + (size_t)tok * DD + d0;
    float v[16];
    float s = 0.f;
#pragma unroll
    for (int j = 0; j < 16; ++j) { v[j] = zr[j] + e[j] * SQRTD; s += v[j]; }
    const float mean = wave_sum(s) * (1.f / 512.f);
    float q = 0.f;
#pragma unroll
    for (int j = 0; j < 16; ++j) { float dx = v[j] - mean; q += dx * dx; }
    const float sd = sqrtf(wave_sum(q) * (1.f / 511.f));
    const float inv = 1.f / (sd + LN_EPS);
    h16* zh = z16hist + ((size_t)t * BB + b) * DD + d0;
#pragma unroll
    for (int j = 0; j < 16; ++j) {
      zr[j] = g1[j] * (v[j] - mean) * inv + b1[j];
      zh[j] = (h16)zr[j];
    }
  }
#pragma unroll
  for (int j = 0; j < 16; ++j) zfin[(size_t)b * DD + d0 + j] = zr[j];
}

// ------------ gate GEMM: gates = sigmoid(scale*(Zhist @ Wt^T + cg)) -------
// M=32768, N=64, K=512 -> 8192 tiles, one per wave.
__global__ void __launch_bounds__(256)
k_gate(const h16* __restrict__ Z, const h16* __restrict__ Wt16,
       const float* __restrict__ cg, float* __restrict__ gates) {
  const int wtile = blockIdx.x * 8 + (threadIdx.x >> 5);
  const int lane = threadIdx.x & 31;
  const int mt = wtile >> 2, nt = wtile & 3;
  v8f acc = {};
  for (int kk = 0; kk < DD; kk += 32) {
    v16h a = load_a(Z + (size_t)mt * 16 * DD, lane, kk, DD);
    v16h bm = load_bfrag(Wt16, lane, nt * 16, kk, DD);
    acc = wmma16(a, bm, acc);
  }
  const int n = nt * 16 + (lane & 15);
  const int m0 = mt * 16 + ((lane >> 4) << 3);
  const float c = cg[n];
#pragma unroll
  for (int r = 0; r < 8; ++r) {
    const float x = SCALE * (acc[r] + c);
    gates[(size_t)(m0 + r) * AA + n] = 1.f / (1.f + __expf(-x));
  }
}

// -------------------------------------------------------- av scan ----------
__global__ void __launch_bounds__(256)
k_avscan(const h16* __restrict__ z16hist, const float* __restrict__ gates,
         const float* __restrict__ n1g, const float* __restrict__ n1b,
         h16* __restrict__ av16) {
  const int w = blockIdx.x * 8 + (threadIdx.x >> 5);  // (b,a) row 0..2047
  const int lane = threadIdx.x & 31;
  const int b = w >> 6, a = w & 63;
  const int d0 = lane << 4;
  float avr[16], g1[16], b1[16];
#pragma unroll
  for (int j = 0; j < 16; ++j) {
    avr[j] = 0.f;
    g1[j] = n1g[d0 + j];
    b1[j] = n1b[d0 + j];
  }
  for (int t = 0; t < SENC; ++t) {
    const float g = gates[((size_t)t * BB + b) * AA + a];
    const h16* z = z16hist + ((size_t)t * BB + b) * DD + d0;
    float v[16];
    float s = 0.f;
#pragma unroll
    for (int j = 0; j < 16; ++j) {
      v[j] = avr[j] + g * ((float)z[j] - avr[j]);
      s += v[j];
    }
    const float mean = wave_sum(s) * (1.f / 512.f);
    float q = 0.f;
#pragma unroll
    for (int j = 0; j < 16; ++j) { float dx = v[j] - mean; q += dx * dx; }
    const float sd = sqrtf(wave_sum(q) * (1.f / 511.f));
    const float inv = 1.f / (sd + LN_EPS);
#pragma unroll
    for (int j = 0; j < 16; ++j) avr[j] = g1[j] * (v[j] - mean) * inv + b1[j];
  }
#pragma unroll
  for (int j = 0; j < 16; ++j) av16[(size_t)w * DD + d0 + j] = (h16)avr[j];
}

// ------------------- generic WMMA tile GEMM: C = A16 @ W^T + bias ----------
template <typename BT, typename OT>
__global__ void __launch_bounds__(256)
k_gemm(const h16* __restrict__ A, const BT* __restrict__ W,
       const float* __restrict__ bias, OT* __restrict__ C,
       int M, int N, int K) {
  const int wtile = blockIdx.x * 8 + (threadIdx.x >> 5);
  const int lane = threadIdx.x & 31;
  const int ntn = N >> 4;
  if (wtile >= (M >> 4) * ntn) return;                // wave-uniform
  const int mt = wtile / ntn, nt = wtile % ntn;
  const int nln = nt * 16 + (lane & 15);
  v8f acc = {};
  for (int kk = 0; kk < K; kk += 32) {
    if (kk + 32 < K)
      __builtin_prefetch(W + (size_t)nln * K + kk + 32, 0, 1);
    v16h a = load_a(A + (size_t)mt * 16 * K, lane, kk, K);
    v16h bm = load_bfrag(W, lane, nt * 16, kk, K);
    acc = wmma16(a, bm, acc);
  }
  const int m0 = mt * 16 + ((lane >> 4) << 3);
  const float bi = bias ? bias[nln] : 0.f;
#pragma unroll
  for (int r = 0; r < 8; ++r) C[(size_t)(m0 + r) * N + nln] = (OT)(acc[r] + bi);
}

// ----------------- decoder attention scan (32 independent waves) ----------
__global__ void __launch_bounds__(32)
k_decoder(const float* __restrict__ zinit, const float* __restrict__ Ktil,
          const float* __restrict__ cb, const float* __restrict__ Vr,
          const float* __restrict__ n2g, const float* __restrict__ n2b,
          h16* __restrict__ zout16) {
  __shared__ float zsh[DD];
  const int b = blockIdx.x;
  const int lane = threadIdx.x;
  const int d0 = lane << 4;
  float zr[16], g2[16], b2[16];
#pragma unroll
  for (int j = 0; j < 16; ++j) {
    zr[j] = zinit[(size_t)b * DD + d0 + j];
    g2[j] = n2g[d0 + j];
    b2[j] = n2b[d0 + j];
  }
  const float* Kb = Ktil + (size_t)b * AA * DD;
  const float* Vb = Vr + (size_t)b * AA * DD;
  const float cb0 = cb[b * AA + lane];
  const float cb1 = cb[b * AA + lane + 32];
  const float* k0 = Kb + (size_t)lane * DD;
  const float* k1 = Kb + (size_t)(lane + 32) * DD;

  for (int t = 0; t < SDEC; ++t) {
#pragma unroll
    for (int j = 0; j < 16; ++j) zsh[d0 + j] = zr[j];
    __syncthreads();
    // att logits for anchors a=lane, a=lane+32
    float s0 = 0.f, s1 = 0.f;
#pragma unroll 4
    for (int k = 0; k < DD; ++k) {
      const float zz = zsh[k];
      s0 += zz * k0[k];
      s1 += zz * k1[k];
    }
    s0 = SCALE * (s0 + cb0);
    s1 = SCALE * (s1 + cb1);
    const float mx = wave_max(fmaxf(s0, s1));
    float e0 = __expf(s0 - mx), e1 = __expf(s1 - mx);
    const float inv = 1.f / wave_sum(e0 + e1);
    e0 *= inv;
    e1 *= inv;
    // dg2 = att @ Vr via lane broadcasts
    float acc[16];
#pragma unroll
    for (int j = 0; j < 16; ++j) acc[j] = 0.f;
    for (int a = 0; a < 32; ++a) {
      const float w0 = __shfl(e0, a);
      const float w1 = __shfl(e1, a);
      const float* v0 = Vb + (size_t)a * DD + d0;
      const float* v1 = Vb + (size_t)(a + 32) * DD + d0;
#pragma unroll
      for (int j = 0; j < 16; ++j) acc[j] += w0 * v0[j] + w1 * v1[j];
    }
    // z = LN(z + dg2)
    float v[16];
    float s = 0.f;
#pragma unroll
    for (int j = 0; j < 16; ++j) { v[j] = zr[j] + acc[j]; s += v[j]; }
    const float mean = wave_sum(s) * (1.f / 512.f);
    float q = 0.f;
#pragma unroll
    for (int j = 0; j < 16; ++j) { float dx = v[j] - mean; q += dx * dx; }
    const float sd = sqrtf(wave_sum(q) * (1.f / 511.f));
    const float ivn = 1.f / (sd + LN_EPS);
#pragma unroll
    for (int j = 0; j < 16; ++j) zr[j] = g2[j] * (v[j] - mean) * ivn + b2[j];
    __syncthreads();
  }
#pragma unroll
  for (int j = 0; j < 16; ++j) zout16[(size_t)b * DD + d0 + j] = (h16)zr[j];
}

// ------------------------------------------------------------ logsoftmax ----
__global__ void __launch_bounds__(256)
k_logsoftmax(const float* __restrict__ logits, float* __restrict__ out) {
  __shared__ float red[256];
  const int b = blockIdx.x, tid = threadIdx.x;
  const float* row = logits + (size_t)b * VOUT;
  float mx = -3.4e38f;
  for (int i = tid; i < VOUT; i += 256) mx = fmaxf(mx, row[i]);
  red[tid] = mx;
  __syncthreads();
  for (int o = 128; o > 0; o >>= 1) {
    if (tid < o) red[tid] = fmaxf(red[tid], red[tid + o]);
    __syncthreads();
  }
  const float m = red[0];
  __syncthreads();
  float s = 0.f;
  for (int i = tid; i < VOUT; i += 256) s += __expf(row[i] - m);
  red[tid] = s;
  __syncthreads();
  for (int o = 128; o > 0; o >>= 1) {
    if (tid < o) red[tid] += red[tid + o];
    __syncthreads();
  }
  const float lse = m + __logf(red[0]);
  for (int i = tid; i < VOUT; i += 256) out[(size_t)b * VOUT + i] = row[i] - lse;
}

// ------------------------------------------------------------------ host ----
static inline void* wsalloc(char* base, size_t& off, size_t bytes) {
  void* p = base + off;
  off = (off + bytes + 255) & ~(size_t)255;
  return p;
}

extern "C" void kernel_launch(void* const* d_in, const int* in_sizes, int n_in,
                              void* d_out, int out_size, void* d_ws,
                              size_t ws_size, hipStream_t stream) {
  const int* ids = (const int*)d_in[0];
  const float* emb_in = (const float*)d_in[2];
  const float* enc_key_W = (const float*)d_in[4];
  const float* enc_Wq = (const float*)d_in[5];
  const float* enc_bq = (const float*)d_in[6];
  const float* enc_Wk = (const float*)d_in[7];
  const float* enc_bk = (const float*)d_in[8];
  const float* n1g = (const float*)d_in[9];
  const float* n1b = (const float*)d_in[10];
  const float* rdr_Wq = (const float*)d_in[12];
  const float* rdr_bq = (const float*)d_in[13];
  const float* rdr_Wk = (const float*)d_in[14];
  const float* rdr_bk = (const float*)d_in[15];
  const float* rdr_Wv = (const float*)d_in[16];
  const float* rdr_bv = (const float*)d_in[17];
  const float* n2g = (const float*)d_in[22];
  const float* n2b = (const float*)d_in[23];
  const float* voc_W = (const float*)d_in[26];
  const float* voc_b = (const float*)d_in[27];
  float* out = (float*)d_out;

  char* ws = (char*)d_ws;
  size_t off = 0;
  h16* WkT16 = (h16*)wsalloc(ws, off, (size_t)DD * DD * 2);
  h16* WqT16 = (h16*)wsalloc(ws, off, (size_t)DD * DD * 2);
  h16* Qa16 = (h16*)wsalloc(ws, off, (size_t)AA * DD * 2);
  float* Qaf = (float*)wsalloc(ws, off, (size_t)AA * DD * 4);
  float* cgate = (float*)wsalloc(ws, off, (size_t)AA * 4);
  h16* Wt16 = (h16*)wsalloc(ws, off, (size_t)AA * DD * 2);
  h16* z16h = (h16*)wsalloc(ws, off, (size_t)SENC * BB * DD * 2);     // 32 MB
  float* gates = (float*)wsalloc(ws, off, (size_t)SENC * BB * AA * 4); // 8 MB
  float* zenc = (float*)wsalloc(ws, off, (size_t)BB * DD * 4);
  h16* av16 = (h16*)wsalloc(ws, off, (size_t)BB * AA * DD * 2);       // 2 MB
  h16* Kr16 = (h16*)wsalloc(ws, off, (size_t)BB * AA * DD * 2);       // 2 MB
  float* Vr = (float*)wsalloc(ws, off, (size_t)BB * AA * DD * 4);     // 4 MB
  float* cb = (float*)wsalloc(ws, off, (size_t)BB * AA * 4);
  float* Ktil = (float*)wsalloc(ws, off, (size_t)BB * AA * DD * 4);   // 4 MB
  h16* zdec16 = (h16*)wsalloc(ws, off, (size_t)BB * DD * 2);
  float* logits = (float*)wsalloc(ws, off, (size_t)BB * VOUT * 4);    // 4 MB

  // ---- prep (one-shot) ----
  k_transpose_to_f16<<<DD * DD / 256, 256, 0, stream>>>(enc_Wk, WkT16);
  k_transpose_to_f16<<<DD * DD / 256, 256, 0, stream>>>(rdr_Wq, WqT16);
  k_qa<<<AA * DD / 256, 256, 0, stream>>>(enc_key_W, enc_Wq, enc_bq, Qa16, Qaf);
  k_cvec<<<1, 64, 0, stream>>>(enc_bk, Qaf, cgate);
  // Wt = Qa @ Wk  : A=Qa16, W=WkT16 (f16), M=64 N=512 K=512
  k_gemm<h16, h16><<<(4 * 32) / 8, 256, 0, stream>>>(
      Qa16, WkT16, (const float*)nullptr, Wt16, AA, DD, DD);

  // ---- encoder ----
  k_enc_zscan<<<BB, 32, 0, stream>>>(ids, emb_in, n1g, n1b, z16h, zenc);
  k_gate<<<(2048 * 4) / 8, 256, 0, stream>>>(z16h, Wt16, cgate, gates);
  k_avscan<<<BB * AA / 8, 256, 0, stream>>>(z16h, gates, n1g, n1b, av16);

  // ---- decoder constants ----
  const int tilesKV = (BB * AA / 16) * (DD / 16);   // 4096
  k_gemm<float, h16><<<tilesKV / 8, 256, 0, stream>>>(
      av16, rdr_Wk, rdr_bk, Kr16, BB * AA, DD, DD);
  k_gemm<float, float><<<tilesKV / 8, 256, 0, stream>>>(
      av16, rdr_Wv, rdr_bv, Vr, BB * AA, DD, DD);
  k_cb<<<BB * AA / 256, 256, 0, stream>>>(rdr_bq, Kr16, cb);
  // K~ = Kr @ Wq : A=Kr16, W=WqT16 (f16)
  k_gemm<h16, float><<<tilesKV / 8, 256, 0, stream>>>(
      Kr16, WqT16, (const float*)nullptr, Ktil, BB * AA, DD, DD);

  // ---- decoder scan ----
  k_decoder<<<BB, 32, 0, stream>>>(zenc, Ktil, cb, Vr, n2g, n2b, zdec16);

  // ---- vocab + log_softmax ----
  const int tilesV = (BB / 16) * (VOUT / 16);       // 4000
  k_gemm<float, float><<<tilesV / 8, 256, 0, stream>>>(
      zdec16, voc_W, voc_b, logits, BB, VOUT, DD);
  k_logsoftmax<<<BB, 256, 0, stream>>>(logits, out);
}
